// MixtralDecoderLayer_15779709845993
// MI455X (gfx1250) — matrix operand order repaired and checked by reading, not compiled, not run on hardware
//
#include <hip/hip_runtime.h>
#include <hip/hip_bf16.h>
#include <math.h>

#define DEVI __device__ __forceinline__

typedef __attribute__((ext_vector_type(16))) __bf16 bf16x16;
typedef __attribute__((ext_vector_type(8)))  float  f32x8;
typedef unsigned short ushort_t;
typedef unsigned int   uint_t;

// POD 16-byte vectors (avoid HIP vector-type ctors inside unions)
struct __align__(16) U128 { uint_t x, y, z, w; };
struct __align__(16) F128 { float  x, y, z, w; };
union Frag32B { U128 q[2]; bf16x16 bf; };   // 32 bytes == v16bf16
static_assert(sizeof(Frag32B) == 32, "frag size");

// CDNA5 async global->LDS path (guarded: falls back to VGPR staging if absent)
#if defined(__has_builtin)
#  if __has_builtin(__builtin_amdgcn_global_load_async_to_lds_b128) && \
      __has_builtin(__builtin_amdgcn_s_wait_asynccnt)
#    define USE_ASYNC_LDS 1
#  endif
#  if __has_builtin(__builtin_amdgcn_cvt_pk_bf16_f32)
#    define USE_CVT_PK 1
#  endif
#endif
#ifdef USE_ASYNC_LDS
typedef int v4i_async __attribute__((vector_size(16)));   // matches builtin param type
#endif

DEVI ushort_t f2bf(float f) {                 // f32 -> bf16 RNE
    uint_t u = __float_as_uint(f);
    u += 0x7FFFu + ((u >> 16) & 1u);
    return (ushort_t)(u >> 16);
}
DEVI uint_t pk2bf(float a, float b) {         // pack bf16(a) | bf16(b)<<16
#ifdef USE_CVT_PK
    auto v2 = __builtin_amdgcn_cvt_pk_bf16_f32(a, b);
    return __builtin_bit_cast(uint_t, v2);
#else
    uint_t ua = __float_as_uint(a);
    ua = (ua + 0x7FFFu + ((ua >> 16) & 1u)) >> 16;
    uint_t ub = __float_as_uint(b);
    ub = (ub + 0x7FFFu + ((ub >> 16) & 1u)) & 0xFFFF0000u;
    return ua | ub;
#endif
}

// Stage one 16-f32 chunk as 16 bf16 (32B) into LDS
DEVI void stageB16(const float* gp, ushort_t* lp) {
    const F128* g = (const F128*)gp;
    F128 f0 = g[0], f1 = g[1], f2 = g[2], f3 = g[3];
    U128 o0, o1;
    o0.x = pk2bf(f0.x, f0.y); o0.y = pk2bf(f0.z, f0.w);
    o0.z = pk2bf(f1.x, f1.y); o0.w = pk2bf(f1.z, f1.w);
    o1.x = pk2bf(f2.x, f2.y); o1.y = pk2bf(f2.z, f2.w);
    o1.z = pk2bf(f3.x, f3.y); o1.w = pk2bf(f3.z, f3.w);
    U128* l = (U128*)lp;
    l[0] = o0; l[1] = o1;
}

// Stage the 64B A-row chunk (async if available)
DEVI void stageA64(const ushort_t* g, ushort_t* l) {
#ifdef USE_ASYNC_LDS
    __builtin_amdgcn_global_load_async_to_lds_b128((v4i_async*)g, (v4i_async*)l, 0, 0);
    __builtin_amdgcn_global_load_async_to_lds_b128((v4i_async*)g, (v4i_async*)l, 16, 0);
    __builtin_amdgcn_global_load_async_to_lds_b128((v4i_async*)g, (v4i_async*)l, 32, 0);
    __builtin_amdgcn_global_load_async_to_lds_b128((v4i_async*)g, (v4i_async*)l, 48, 0);
#else
    const U128* gq = (const U128*)g;
    U128* lq = (U128*)l;
    lq[0] = gq[0]; lq[1] = gq[1]; lq[2] = gq[2]; lq[3] = gq[3];
#endif
}
DEVI void waitAsync() {
#ifdef USE_ASYNC_LDS
    __builtin_amdgcn_s_wait_asynccnt(0);
#endif
}

// ---------------------------------------------------------------------------
// RMSNorm: one block per token row of 1024. Writes bf16 (and optional f32).
// ---------------------------------------------------------------------------
__global__ __launch_bounds__(256) void rmsnorm_kernel(
    const float* __restrict__ X, const float* __restrict__ w,
    ushort_t* __restrict__ outB, float* __restrict__ outF)
{
    const int t = blockIdx.x;
    const float* x = X + (size_t)t * 1024;
    float v[4]; float ss = 0.f;
#pragma unroll
    for (int i = 0; i < 4; ++i) { v[i] = x[threadIdx.x + i * 256]; ss += v[i] * v[i]; }
    __shared__ float red[256];
    red[threadIdx.x] = ss; __syncthreads();
    for (int off = 128; off > 0; off >>= 1) {
        if (threadIdx.x < off) red[threadIdx.x] += red[threadIdx.x + off];
        __syncthreads();
    }
    const float rms = rsqrtf(red[0] * (1.0f / 1024.0f) + 1e-5f);
#pragma unroll
    for (int i = 0; i < 4; ++i) {
        const int c = threadIdx.x + i * 256;
        const float y = v[i] * rms * w[c];
        outB[(size_t)t * 1024 + c] = f2bf(y);
        if (outF) outF[(size_t)t * 1024 + c] = y;
    }
}

// ---------------------------------------------------------------------------
// WMMA GEMM:  C[M,N] = A_bf16[M,K] * W_f32[N,K]^T  (W converted bf16 in LDS)
// Tile: BM=128, BN=64, BK=64. 256 threads = 8 waves (4x2), 32x32 per wave.
// mode 0: outF = acc (+aux residual)            (slot-indexed out)
// mode 1: outB = bf16(acc)
// mode 3: outF[gathered] += acc * scale[slot]
// ---------------------------------------------------------------------------
__global__ __launch_bounds__(256) void gemm_bf16_wmma(
    const ushort_t* __restrict__ A, const float* __restrict__ W,
    int M, int N, int K,
    const int* __restrict__ rowmap, const int* __restrict__ cntPtr,
    int aGather, int outGather, int mode,
    float* __restrict__ outF, ushort_t* __restrict__ outB,
    const float* __restrict__ aux, const float* __restrict__ scale)
{
    const int Meff = cntPtr ? *cntPtr : M;
    const int m0 = blockIdx.x * 128;
    if (m0 >= Meff) return;
    const int n0 = blockIdx.y * 64;

    __shared__ ushort_t sA[128][72];   // BK=64 + 8 pad
    __shared__ ushort_t sB[64][72];

    const int tid  = threadIdx.x;
    const int lane = tid & 31;
    const int wm   = (tid >> 5) & 3;     // wave m 0..3
    const int wn   = (tid >> 5) >> 2;    // wave n 0..1
    const int half = lane >> 4;
    const int l16  = lane & 15;

    f32x8 acc[2][2];
#pragma unroll
    for (int i = 0; i < 2; ++i)
#pragma unroll
        for (int j = 0; j < 2; ++j)
#pragma unroll
            for (int r = 0; r < 8; ++r) acc[i][j][r] = 0.f;

    // A staging: 2 threads/row, 32 ush (64B) each. B: 4 threads/row, 16 f32 each.
    const int arow = tid >> 1, ah = tid & 1;
    const int brow = tid >> 2, bq = tid & 3;
    int am = m0 + arow; if (am >= Meff) am = Meff - 1;   // clamp: dup rows, epilogue skips
    const int atok = aGather ? rowmap[am] : am;
    const ushort_t* aRow = A + (size_t)atok * K + ah * 32;
    const float*    bRow = W + (size_t)(n0 + brow) * K + bq * 16;

    for (int k0 = 0; k0 < K; k0 += 64) {
        stageA64(aRow + k0, &sA[arow][ah * 32]);
        stageB16(bRow + k0, &sB[brow][bq * 16]);
        if (k0 + 64 < K) __builtin_prefetch((const void*)(bRow + k0 + 64), 0, 1);
        waitAsync();
        __syncthreads();

        // ---- fragments + WMMA (2 K-chunks of 32) ----
#pragma unroll
        for (int kc = 0; kc < 2; ++kc) {
            Frag32B fa[2], fb[2];
#pragma unroll
            for (int i = 0; i < 2; ++i) {
                const ushort_t* r = &sA[wm * 32 + i * 16 + l16][kc * 32];
                fa[i].q[0] = *(const U128*)(r + half * 16);
                fa[i].q[1] = *(const U128*)(r + half * 16 + 8);
            }
#pragma unroll
            for (int j = 0; j < 2; ++j) {
                const ushort_t* r = &sB[wn * 32 + j * 16 + l16][kc * 32];
                fb[j].q[0] = *(const U128*)(r + half * 16);
                fb[j].q[1] = *(const U128*)(r + half * 16 + 8);
            }
#pragma unroll
            for (int i = 0; i < 2; ++i)
#pragma unroll
                for (int j = 0; j < 2; ++j)
                    acc[i][j] = __builtin_amdgcn_wmma_f32_16x16x32_bf16(
                        false, fa[i].bf, false, fb[j].bf, (short)0, acc[i][j], false, false);
        }
        __syncthreads();
    }

    // ---- epilogue ----
#pragma unroll
    for (int i = 0; i < 2; ++i)
#pragma unroll
        for (int j = 0; j < 2; ++j)
#pragma unroll
            for (int r = 0; r < 8; ++r) {
                const int mi = m0 + wm * 32 + i * 16 + r + 8 * half;
                if (mi >= Meff) continue;
                const int n = n0 + wn * 32 + j * 16 + l16;
                const float a = acc[i][j][r];
                const size_t slotIdx = (size_t)mi * N + n;
                const int orow = outGather ? rowmap[mi] : mi;
                const size_t outIdx = (size_t)orow * N + n;
                if (mode == 0) {
                    outF[outIdx] = a + (aux ? aux[slotIdx] : 0.f);
                } else if (mode == 1) {
                    outB[outIdx] = f2bf(a);
                } else {
                    outF[outIdx] += a * scale[mi];
                }
            }
}

// ---------------------------------------------------------------------------
// Fused MoE up-projection:  hE = bf16( silu(A@W1^T) * (A@W3^T) ), A gathered.
// Same tiling as gemm_bf16_wmma; two B tiles, dual accumulators, 16 WMMA/iter.
// ---------------------------------------------------------------------------
__global__ __launch_bounds__(256) void gemm_dual_silu_wmma(
    const ushort_t* __restrict__ A,
    const float* __restrict__ W1, const float* __restrict__ W3,
    int M, int N, int K,
    const int* __restrict__ rowmap, const int* __restrict__ cntPtr,
    ushort_t* __restrict__ outB)
{
    const int Meff = cntPtr ? *cntPtr : M;
    const int m0 = blockIdx.x * 128;
    if (m0 >= Meff) return;
    const int n0 = blockIdx.y * 64;

    __shared__ ushort_t sA[128][72];
    __shared__ ushort_t sB1[64][72];
    __shared__ ushort_t sB3[64][72];

    const int tid  = threadIdx.x;
    const int lane = tid & 31;
    const int wm   = (tid >> 5) & 3;
    const int wn   = (tid >> 5) >> 2;
    const int half = lane >> 4;
    const int l16  = lane & 15;

    f32x8 acc1[2][2], acc3[2][2];
#pragma unroll
    for (int i = 0; i < 2; ++i)
#pragma unroll
        for (int j = 0; j < 2; ++j)
#pragma unroll
            for (int r = 0; r < 8; ++r) { acc1[i][j][r] = 0.f; acc3[i][j][r] = 0.f; }

    const int arow = tid >> 1, ah = tid & 1;
    const int brow = tid >> 2, bq = tid & 3;
    int am = m0 + arow; if (am >= Meff) am = Meff - 1;
    const int atok = rowmap[am];
    const ushort_t* aRow  = A  + (size_t)atok * K + ah * 32;
    const float*    b1Row = W1 + (size_t)(n0 + brow) * K + bq * 16;
    const float*    b3Row = W3 + (size_t)(n0 + brow) * K + bq * 16;

    for (int k0 = 0; k0 < K; k0 += 64) {
        stageA64(aRow + k0, &sA[arow][ah * 32]);
        stageB16(b1Row + k0, &sB1[brow][bq * 16]);
        stageB16(b3Row + k0, &sB3[brow][bq * 16]);
        if (k0 + 64 < K) {
            __builtin_prefetch((const void*)(b1Row + k0 + 64), 0, 1);
            __builtin_prefetch((const void*)(b3Row + k0 + 64), 0, 1);
        }
        waitAsync();
        __syncthreads();

#pragma unroll
        for (int kc = 0; kc < 2; ++kc) {
            Frag32B fa[2], fb1[2], fb3[2];
#pragma unroll
            for (int i = 0; i < 2; ++i) {
                const ushort_t* r = &sA[wm * 32 + i * 16 + l16][kc * 32];
                fa[i].q[0] = *(const U128*)(r + half * 16);
                fa[i].q[1] = *(const U128*)(r + half * 16 + 8);
            }
#pragma unroll
            for (int j = 0; j < 2; ++j) {
                const ushort_t* r1 = &sB1[wn * 32 + j * 16 + l16][kc * 32];
                fb1[j].q[0] = *(const U128*)(r1 + half * 16);
                fb1[j].q[1] = *(const U128*)(r1 + half * 16 + 8);
                const ushort_t* r3 = &sB3[wn * 32 + j * 16 + l16][kc * 32];
                fb3[j].q[0] = *(const U128*)(r3 + half * 16);
                fb3[j].q[1] = *(const U128*)(r3 + half * 16 + 8);
            }
#pragma unroll
            for (int i = 0; i < 2; ++i)
#pragma unroll
                for (int j = 0; j < 2; ++j) {
                    acc1[i][j] = __builtin_amdgcn_wmma_f32_16x16x32_bf16(
                        false, fa[i].bf, false, fb1[j].bf, (short)0, acc1[i][j], false, false);
                    acc3[i][j] = __builtin_amdgcn_wmma_f32_16x16x32_bf16(
                        false, fa[i].bf, false, fb3[j].bf, (short)0, acc3[i][j], false, false);
                }
        }
        __syncthreads();
    }

#pragma unroll
    for (int i = 0; i < 2; ++i)
#pragma unroll
        for (int j = 0; j < 2; ++j)
#pragma unroll
            for (int r = 0; r < 8; ++r) {
                const int mi = m0 + wm * 32 + i * 16 + r + 8 * half;
                if (mi >= Meff) continue;
                const int n = n0 + wn * 32 + j * 16 + l16;
                const float a1 = acc1[i][j][r];
                const float a3 = acc3[i][j][r];
                const float s = a1 / (1.f + __expf(-a1));
                outB[(size_t)mi * N + n] = f2bf(s * a3);
            }
}

// ---------------------------------------------------------------------------
// V transpose (LDS-tiled, coalesced both sides):
// [B,S,NH,D] (==[T,1024]) -> VT [B,NH,D,S].  Block: one (b,h), 64 s-rows.
// ---------------------------------------------------------------------------
__global__ __launch_bounds__(256) void vtrans_kernel(
    const ushort_t* __restrict__ V, ushort_t* __restrict__ VT)
{
    __shared__ ushort_t tile[64][72];
    const int bh = blockIdx.x;                 // b*16 + h
    const int b = bh >> 4, h = bh & 15;
    const int s0 = blockIdx.y * 64;
    const int r = threadIdx.x >> 2, q = threadIdx.x & 3;
    {
        const U128* src = (const U128*)(V + ((size_t)(b * 2048 + s0 + r)) * 1024 + h * 64 + q * 16);
        *(U128*)&tile[r][q * 16]     = src[0];
        *(U128*)&tile[r][q * 16 + 8] = src[1];
    }
    __syncthreads();
    {
        const int sb = q * 16;                 // r is now the d index
        U128 u0, u1;
        u0.x = (uint_t)tile[sb + 0][r] | ((uint_t)tile[sb + 1][r] << 16);
        u0.y = (uint_t)tile[sb + 2][r] | ((uint_t)tile[sb + 3][r] << 16);
        u0.z = (uint_t)tile[sb + 4][r] | ((uint_t)tile[sb + 5][r] << 16);
        u0.w = (uint_t)tile[sb + 6][r] | ((uint_t)tile[sb + 7][r] << 16);
        u1.x = (uint_t)tile[sb + 8][r] | ((uint_t)tile[sb + 9][r] << 16);
        u1.y = (uint_t)tile[sb + 10][r] | ((uint_t)tile[sb + 11][r] << 16);
        u1.z = (uint_t)tile[sb + 12][r] | ((uint_t)tile[sb + 13][r] << 16);
        u1.w = (uint_t)tile[sb + 14][r] | ((uint_t)tile[sb + 15][r] << 16);
        ushort_t* dst = VT + ((size_t)(bh * 64 + r)) * 2048 + s0 + sb;
        ((U128*)dst)[0] = u0; ((U128*)dst)[1] = u1;
    }
}

// ---------------------------------------------------------------------------
// Flash attention, WMMA bf16. Block = 128 thr (4 waves), 64 Q rows/block.
// Q,K: bf16 [B,S,NH,D]; VT: bf16 [B,NH,D,S]; O: bf16 [B,S,NH,D].
// ---------------------------------------------------------------------------
__global__ __launch_bounds__(128) void flash_attn_kernel(
    const ushort_t* __restrict__ Q, const ushort_t* __restrict__ Kc,
    const ushort_t* __restrict__ VT, ushort_t* __restrict__ O)
{
    const int b = blockIdx.x >> 4, h = blockIdx.x & 15;
    const int q0 = blockIdx.y * 64;
    const int lane = threadIdx.x & 31, wave = threadIdx.x >> 5;
    const int half = lane >> 4, l16 = lane & 15;
    const int qrowBase = q0 + wave * 16;

    __shared__ ushort_t P[4][16][40];

    Frag32B aq[2];
    {
        const ushort_t* qp = Q + ((size_t)(b * 2048 + qrowBase + l16)) * 1024 + h * 64;
#pragma unroll
        for (int kc = 0; kc < 2; ++kc) {
            aq[kc].q[0] = *(const U128*)(qp + kc * 32 + half * 16);
            aq[kc].q[1] = *(const U128*)(qp + kc * 32 + half * 16 + 8);
        }
    }
    float mrow[8], lrow[8];
#pragma unroll
    for (int r = 0; r < 8; ++r) { mrow[r] = -3.0e38f; lrow[r] = 0.f; }
    f32x8 o[4];
#pragma unroll
    for (int dt = 0; dt < 4; ++dt)
#pragma unroll
        for (int r = 0; r < 8; ++r) o[dt][r] = 0.f;

    for (int kb = 0; kb < 2048; kb += 32) {
        // scores S = Q K^T  (16 x 32)
        f32x8 s[2];
#pragma unroll
        for (int jn = 0; jn < 2; ++jn) {
#pragma unroll
            for (int r = 0; r < 8; ++r) s[jn][r] = 0.f;
            const int key = kb + jn * 16 + l16;
            const ushort_t* kp = Kc + ((size_t)(b * 2048 + key)) * 1024 + h * 64;
#pragma unroll
            for (int kc = 0; kc < 2; ++kc) {
                Frag32B bk;
                bk.q[0] = *(const U128*)(kp + kc * 32 + half * 16);
                bk.q[1] = *(const U128*)(kp + kc * 32 + half * 16 + 8);
                s[jn] = __builtin_amdgcn_wmma_f32_16x16x32_bf16(
                    false, aq[kc].bf, false, bk.bf, (short)0, s[jn], false, false);
            }
        }
        // online softmax (row stats via shfl_xor within 16-lane groups)
        float alpha[8];
#pragma unroll
        for (int r = 0; r < 8; ++r) {
            float v0 = s[0][r] * 0.125f, v1 = s[1][r] * 0.125f;
            float mx = fmaxf(v0, v1);
#pragma unroll
            for (int d = 1; d < 16; d <<= 1) mx = fmaxf(mx, __shfl_xor(mx, d, 32));
            const float nm = fmaxf(mrow[r], mx);
            alpha[r] = __expf(mrow[r] - nm);
            const float p0 = __expf(v0 - nm), p1 = __expf(v1 - nm);
            s[0][r] = p0; s[1][r] = p1;
            float rs = p0 + p1;
#pragma unroll
            for (int d = 1; d < 16; d <<= 1) rs += __shfl_xor(rs, d, 32);
            lrow[r] = lrow[r] * alpha[r] + rs;
            mrow[r] = nm;
        }
#pragma unroll
        for (int dt = 0; dt < 4; ++dt)
#pragma unroll
            for (int r = 0; r < 8; ++r) o[dt][r] *= alpha[r];

        // P (C-layout) -> LDS -> A-layout fragment
#pragma unroll
        for (int r = 0; r < 8; ++r) {
            P[wave][r + 8 * half][l16]      = f2bf(s[0][r]);
            P[wave][r + 8 * half][16 + l16] = f2bf(s[1][r]);
        }
        __syncthreads();
        Frag32B ap;
        {
            const ushort_t* pp = &P[wave][l16][0];
            ap.q[0] = *(const U128*)(pp + half * 16);
            ap.q[1] = *(const U128*)(pp + half * 16 + 8);
        }
        // O += P V  (V^T rows are contiguous over keys)
#pragma unroll
        for (int dt = 0; dt < 4; ++dt) {
            const int dcol = dt * 16 + l16;
            const ushort_t* vp = VT + ((size_t)((b * 16 + h) * 64 + dcol)) * 2048 + kb + half * 16;
            Frag32B bv;
            bv.q[0] = *(const U128*)(vp);
            bv.q[1] = *(const U128*)(vp + 8);
            o[dt] = __builtin_amdgcn_wmma_f32_16x16x32_bf16(
                false, ap.bf, false, bv.bf, (short)0, o[dt], false, false);
        }
        __syncthreads();
    }
#pragma unroll
    for (int dt = 0; dt < 4; ++dt)
#pragma unroll
        for (int r = 0; r < 8; ++r) {
            const int qr = qrowBase + r + 8 * half;
            const float val = o[dt][r] / lrow[r];
            O[((size_t)(b * 2048 + qr)) * 1024 + h * 64 + dt * 16 + l16] = f2bf(val);
        }
}

// ---------------------------------------------------------------------------
// Gate logits: one block per token, wave w computes expert w (f32 dot of 1024)
// ---------------------------------------------------------------------------
__global__ __launch_bounds__(256) void gate_kernel(
    const float* __restrict__ Xn, const float* __restrict__ Gw, float* __restrict__ logits)
{
    const int t = blockIdx.x;
    const int e = threadIdx.x >> 5, lane = threadIdx.x & 31;
    const float* x = Xn + (size_t)t * 1024;
    const float* g = Gw + (size_t)e * 1024;
    float s = 0.f;
    for (int k = lane; k < 1024; k += 32) s += x[k] * g[k];
#pragma unroll
    for (int d = 1; d < 32; d <<= 1) s += __shfl_xor(s, d, 32);
    if (lane == 0) logits[t * 8 + e] = s;
}

__global__ void zero8_kernel(int* p) { if (threadIdx.x < 8) p[threadIdx.x] = 0; }

// top-2 routing + softmax + atomic compaction into per-expert token lists
__global__ __launch_bounds__(256) void routing_kernel(
    const float* __restrict__ logits, int* __restrict__ cnt,
    int* __restrict__ list, float* __restrict__ wts)
{
    const int t = blockIdx.x * 256 + threadIdx.x;
    if (t >= 4096) return;
    float best = -3.0e38f, second = -3.0e38f; int bi = 0, si = 0;
#pragma unroll
    for (int e = 0; e < 8; ++e) {
        const float v = logits[t * 8 + e];
        if (v > best) { second = best; si = bi; best = v; bi = e; }
        else if (v > second) { second = v; si = e; }
    }
    const float p1 = __expf(second - best);
    const float inv = 1.f / (1.f + p1);
    const int s0 = atomicAdd(&cnt[bi], 1);
    list[bi * 4096 + s0] = t; wts[bi * 4096 + s0] = inv;
    const int s1 = atomicAdd(&cnt[si], 1);
    list[si * 4096 + s1] = t; wts[si * 4096 + s1] = p1 * inv;
}

__global__ __launch_bounds__(256) void copy_kernel(
    const float* __restrict__ src, float* __restrict__ dst, size_t n)
{
    const size_t i = (size_t)blockIdx.x * 256 + threadIdx.x;
    if (i < n) dst[i] = src[i];
}

// ---------------------------------------------------------------------------
extern "C" void kernel_launch(void* const* d_in, const int* in_sizes, int n_in,
                              void* d_out, int out_size, void* d_ws, size_t ws_size,
                              hipStream_t stream)
{
    (void)in_sizes; (void)n_in; (void)out_size; (void)ws_size;
    const float* x    = (const float*)d_in[0];
    const float* ln1w = (const float*)d_in[1];
    const float* ln2w = (const float*)d_in[2];
    const float* wq   = (const float*)d_in[3];
    const float* wk   = (const float*)d_in[4];
    const float* wv   = (const float*)d_in[5];
    const float* wo   = (const float*)d_in[6];
    const float* gw   = (const float*)d_in[7];
    const float* w1   = (const float*)d_in[8];
    const float* w2   = (const float*)d_in[9];
    const float* w3   = (const float*)d_in[10];
    float* out = (float*)d_out;

    const int T = 4096, H = 1024, I = 2048;

    char* p = (char*)d_ws;
    auto carve = [&](size_t bytes) -> char* {
        char* r = p; p += (bytes + 255) & ~(size_t)255; return r;
    };
    ushort_t* xn1  = (ushort_t*)carve((size_t)T * H * 2);
    ushort_t* qb   = (ushort_t*)carve((size_t)T * H * 2);
    ushort_t* kb   = (ushort_t*)carve((size_t)T * H * 2);
    ushort_t* vb   = (ushort_t*)carve((size_t)T * H * 2);
    ushort_t* vtb  = (ushort_t*)carve((size_t)T * H * 2);
    ushort_t* attn = (ushort_t*)carve((size_t)T * H * 2);
    float*    h1   = (float*)   carve((size_t)T * H * 4);
    ushort_t* xn2  = (ushort_t*)carve((size_t)T * H * 2);
    float*    xn2f = (float*)   carve((size_t)T * H * 4);
    float*    lgts = (float*)   carve((size_t)T * 8 * 4);
    int*      cnt  = (int*)     carve(256);
    int*      list = (int*)     carve((size_t)8 * T * 4);
    float*    wts  = (float*)   carve((size_t)8 * T * 4);
    ushort_t* hE   = (ushort_t*)carve((size_t)T * I * 2);

    const dim3 gHH(T / 128, H / 64);   // [T,H] GEMMs
    const dim3 gTI(T / 128, I / 64);   // expert up-proj (worst-case M=T)
    const dim3 gTH(T / 128, H / 64);   // expert down-proj

    // --- attention half ---
    rmsnorm_kernel<<<T, 256, 0, stream>>>(x, ln1w, xn1, nullptr);
    gemm_bf16_wmma<<<gHH, 256, 0, stream>>>(xn1, wq, T, H, H, nullptr, nullptr, 0, 0, 1,
                                            nullptr, qb, nullptr, nullptr);
    gemm_bf16_wmma<<<gHH, 256, 0, stream>>>(xn1, wk, T, H, H, nullptr, nullptr, 0, 0, 1,
                                            nullptr, kb, nullptr, nullptr);
    gemm_bf16_wmma<<<gHH, 256, 0, stream>>>(xn1, wv, T, H, H, nullptr, nullptr, 0, 0, 1,
                                            nullptr, vb, nullptr, nullptr);
    vtrans_kernel<<<dim3(32, 32), 256, 0, stream>>>(vb, vtb);
    flash_attn_kernel<<<dim3(32, 32), 128, 0, stream>>>(qb, kb, vtb, attn);
    gemm_bf16_wmma<<<gHH, 256, 0, stream>>>(attn, wo, T, H, H, nullptr, nullptr, 0, 0, 0,
                                            h1, nullptr, x, nullptr);   // + residual x

    // --- MoE half ---
    rmsnorm_kernel<<<T, 256, 0, stream>>>(h1, ln2w, xn2, xn2f);
    gate_kernel<<<T, 256, 0, stream>>>(xn2f, gw, lgts);
    zero8_kernel<<<1, 8, 0, stream>>>(cnt);
    routing_kernel<<<T / 256, 256, 0, stream>>>(lgts, cnt, list, wts);
    copy_kernel<<<(T * H) / 256, 256, 0, stream>>>(h1, out, (size_t)T * H);

    for (int e = 0; e < 8; ++e) {
        const int*   rm = list + e * T;
        const float* sc = wts + e * T;
        const int*   ce = cnt + e;
        // hE = bf16( silu(gather(xn2)@w1[e]^T) * (gather(xn2)@w3[e]^T) )  [fused]
        gemm_dual_silu_wmma<<<gTI, 256, 0, stream>>>(xn2,
                                                     w1 + (size_t)e * I * H,
                                                     w3 + (size_t)e * I * H,
                                                     T, I, H, rm, ce, hE);
        // out[token] += (hE @ w2[e]^T) * wts     (scatter-accumulate)
        gemm_bf16_wmma<<<gTH, 256, 0, stream>>>(hE, w2 + (size_t)e * H * I, T, H, I,
                                                rm, ce, 0, 1, 3, out, nullptr, nullptr, sc);
    }
}
